// ElectronicSchnet_76940044140594
// MI455X (gfx1250) — matrix operand order repaired and checked by reading, not compiled, not run on hardware
//
#include <hip/hip_runtime.h>
#include <hip/hip_bf16.h>

// ---------------------------------------------------------------------------
// ElectronicSchnet on MI455X (gfx1250), wave32, WMMA bf16.
// One workgroup (256 threads = 8 waves) per batch element. All 3 layers fused;
// per-batch state in LDS; every GEMM on v_wmma_f32_16x16x32_bf16.
// Round 3: hoisted B fragments in Phases A/D, v_perm-packed strided loads.
// ---------------------------------------------------------------------------

#define BB     1024
#define NN     32
#define AA     4
#define BASIS  32
#define MIDW   64
#define KE     128     // KERNEL == EMBED == MID_G == 128
#define LL     3

typedef __attribute__((ext_vector_type(16))) __bf16 v16bf;
typedef __attribute__((ext_vector_type(8)))  float  v8f;

// truncate two f32 -> packed bf16 pair in one v_perm_b32
static __device__ __forceinline__ unsigned pkbf(float lo, float hi) {
    return __builtin_amdgcn_perm(__builtin_bit_cast(unsigned, hi),
                                 __builtin_bit_cast(unsigned, lo),
                                 0x07060302u);
}

static __device__ __forceinline__ __bf16 f2bf(float f) {  // truncate
    unsigned u = __builtin_bit_cast(unsigned, f);
    return __builtin_bit_cast(__bf16, (unsigned short)(u >> 16));
}

// softplus(x) - ln2, stable, fast-math transcendentals
static __device__ __forceinline__ float ssp(float x) {
    float ax = fabsf(x);
    return fmaxf(x, 0.0f) + __logf(1.0f + __expf(-ax)) - 0.6931471805599453f;
}

// ---- WMMA fragment loaders (CDNA5 16x16x32 bf16 layouts, ISA 7.12.2) ------
// A (16x32): lane m=lane&15 holds row m; kb=8*(lane>>4);
//   vgprs 0..3 -> K=kb..kb+7 ; vgprs 4..7 -> K=kb+16..kb+23
static __device__ __forceinline__ v16bf load_a_f32(const float* p, int ld, int lane) {
    int m  = lane & 15;
    int kb = (lane >> 4) * 8;
    const float* row = p + m * ld + kb;
    float4 q0 = *(const float4*)(row);
    float4 q1 = *(const float4*)(row + 4);
    float4 q2 = *(const float4*)(row + 16);
    float4 q3 = *(const float4*)(row + 20);
    union { v16bf v; unsigned u[8]; } r;
    r.u[0] = pkbf(q0.x, q0.y); r.u[1] = pkbf(q0.z, q0.w);
    r.u[2] = pkbf(q1.x, q1.y); r.u[3] = pkbf(q1.z, q1.w);
    r.u[4] = pkbf(q2.x, q2.y); r.u[5] = pkbf(q2.z, q2.w);
    r.u[6] = pkbf(q3.x, q3.y); r.u[7] = pkbf(q3.z, q3.w);
    return r.v;
}

// bf16 source: two contiguous 16-byte groups -> 2x b128 loads, no VALU
static __device__ __forceinline__ v16bf load_a_bf16(const __bf16* p, int ld, int lane) {
    int m  = lane & 15;
    int kb = (lane >> 4) * 8;
    const __bf16* row = p + m * ld + kb;
    union { v16bf v; uint4 q[2]; } r;
    r.q[0] = *(const uint4*)(row);
    r.q[1] = *(const uint4*)(row + 16);
    return r.v;
}

// B (32x16): lane holds column n=lane&15; kb=8*(lane>>4); K=kb..kb+7, kb+16..kb+23.
// k-adjacent element pairs share a VGPR -> one v_perm_b32 per pair.
static __device__ __forceinline__ v16bf load_b_f32(const float* p, int ldn, int lane) {
    int n  = lane & 15;
    int kb = (lane >> 4) * 8;
    const float* q = p + kb * ldn + n;
    union { v16bf v; unsigned u[8]; } r;
#pragma unroll
    for (int t = 0; t < 8; ++t)
        r.u[t] = pkbf(q[(2 * t + (t >= 4 ? 8 : 0)) * ldn], q[(2 * t + 1 + (t >= 4 ? 8 : 0)) * ldn]);
    return r.v;
}

static __device__ __forceinline__ v8f wmma_bf16(v16bf a, v16bf b, v8f c) {
    return __builtin_amdgcn_wmma_f32_16x16x32_bf16(
        false, a, false, b, (short)0, c, false, false);
}

static __device__ __forceinline__ v8f zero8() {
    v8f z = {0.f, 0.f, 0.f, 0.f, 0.f, 0.f, 0.f, 0.f};
    return z;
}

__global__ __launch_bounds__(256, 2)
void ElectronicSchnet_76940044140594_kernel(
    const float* __restrict__ eelec,   // (B, N, N, BASIS)
    const float* __restrict__ enuc,    // (B, N, A, BASIS)
    const float* __restrict__ X,       // (N, EMBED)
    const float* __restrict__ Y,       // (A, KERNEL)
    const float* __restrict__ Ws1, const float* __restrict__ bs1,
    const float* __restrict__ Ws2, const float* __restrict__ bs2,
    const float* __restrict__ Wh,
    const float* __restrict__ Wg1, const float* __restrict__ bg1,
    const float* __restrict__ Wg2, const float* __restrict__ bg2,
    float* __restrict__ out)           // (B, N, EMBED)
{
    __shared__ float  s_x [NN * KE];        // 16 KB current embeddings (f32)
    __shared__ float  s_hx[NN * KE];        // 16 KB hx = x @ Wh (f32)
    __shared__ float  s_z [NN * KE];        // 16 KB z accumulator (f32, ds_add)
    __shared__ __bf16 s_hbuf[2][32 * MIDW]; //  8 KB double-buffered MLP hidden
    __bf16* s_g = &s_hbuf[0][0];            //  aliased gating hidden (32x128)

    const int tid  = threadIdx.x;
    const int lane = tid & 31;
    const int wv   = tid >> 5;
    const int b    = blockIdx.x;
    const int lh   = lane >> 4;             // lane half -> +8 row offset
    const int nl   = lane & 15;

    const int mt1  = wv >> 2, nt1 = wv & 3; // GEMM1 tile (edge MLP stage 1)
    const int col1 = nt1 * 16 + nl;         // GEMM1 output column
    const int col2 = wv * 16 + nl;          // GEMM2 / 128-wide output column

    // Y columns for the nuclear reduction (layer-invariant)
    float yv[AA];
#pragma unroll
    for (int a = 0; a < AA; ++a) yv[a] = Y[a * KE + col2];

    for (int t = tid; t < NN * KE; t += 256) s_x[t] = X[t];
    __syncthreads();

#pragma unroll 1
    for (int layer = 0; layer < LL; ++layer) {
        const float* ws1  = Ws1 + layer * BASIS * MIDW;
        const float* ws2  = Ws2 + layer * MIDW * KE;
        const float* wh   = Wh  + layer * KE * KE;
        const float* wg1  = Wg1 + layer * KE * KE;
        const float* wg2  = Wg2 + layer * KE * KE;

        // ---- per-wave register-cached operands for the edge MLP ------------
        v16bf b1  = load_b_f32(ws1 + nt1 * 16, MIDW, lane);
        v16bf b2a = load_b_f32(ws2 + wv * 16, KE, lane);            // kt=0
        v16bf b2b = load_b_f32(ws2 + 32 * KE + wv * 16, KE, lane);  // kt=1
        const float bias1 = bs1[layer * MIDW + col1];
        const float bias2 = bs2[layer * KE + col2];
        const float biag1 = bg1[layer * KE + col2];
        const float biag2 = bg2[layer * KE + col2];

        // -------- Phase A: hx = x @ Wh  (32x128, K=128) ---------------------
        {
            v16bf bw[4];
#pragma unroll
            for (int kt = 0; kt < 4; ++kt)
                bw[kt] = load_b_f32(wh + kt * 32 * KE + wv * 16, KE, lane);
#pragma unroll
            for (int r = 0; r < 2; ++r) {
                v8f acc = zero8();
#pragma unroll
                for (int kt = 0; kt < 4; ++kt)
                    acc = wmma_bf16(load_a_f32(s_x + r * 16 * KE + kt * 32, KE, lane),
                                    bw[kt], acc);
#pragma unroll
                for (int g = 0; g < 8; ++g)
                    s_hx[(r * 16 + g + 8 * lh) * KE + col2] = acc[g];
            }
        }
        for (int t = tid; t < NN * KE; t += 256) s_z[t] = 0.f;
        __syncthreads();

        // -------- Phase B: z_nuc (4 chunks of 32 nuclear-edge rows) ---------
#pragma unroll 1
        for (int c = 0; c < 4; ++c) {
            const float* Ech = enuc + (size_t)b * (NN * AA * BASIS) + c * 32 * BASIS;
            {   // GEMM1: ssp(E @ Ws1 + bs1) -> s_hbuf[0] (bf16)
                v16bf a = load_a_f32(Ech + mt1 * 16 * BASIS, BASIS, lane);
                v8f acc = wmma_bf16(a, b1, zero8());
#pragma unroll
                for (int g = 0; g < 8; ++g)
                    s_hbuf[0][(mt1 * 16 + g + 8 * lh) * MIDW + col1] =
                        f2bf(ssp(acc[g] + bias1));
            }
            __syncthreads();
            // GEMM2 + reduce against Y (rows r = n*A + a)
#pragma unroll
            for (int r = 0; r < 2; ++r) {
                v8f acc = zero8();
                acc = wmma_bf16(load_a_bf16(s_hbuf[0] + r * 16 * MIDW,      MIDW, lane), b2a, acc);
                acc = wmma_bf16(load_a_bf16(s_hbuf[0] + r * 16 * MIDW + 32, MIDW, lane), b2b, acc);
                // ai = g&3 (compile-time); n advances by 1 between g<4 and g>=4
                float s0 = 0.f, s1 = 0.f;
#pragma unroll
                for (int g = 0; g < 4; ++g) {
                    s0 += (acc[g]     + bias2) * yv[g];
                    s1 += (acc[g + 4] + bias2) * yv[g];
                }
                int n0 = c * 8 + r * 4 + 2 * lh;
                atomicAdd(&s_z[n0 * KE + col2], s0);
                atomicAdd(&s_z[(n0 + 1) * KE + col2], s1);
            }
            __syncthreads();
        }

        // -------- Phase C: z_elec, software-pipelined over 32 chunks --------
        const float* Eb = eelec + (size_t)b * (NN * NN * BASIS);
        {   // prologue: GEMM1 of chunk 0 into buffer 0
            v16bf a = load_a_f32(Eb + mt1 * 16 * BASIS, BASIS, lane);
            v8f acc = wmma_bf16(a, b1, zero8());
#pragma unroll
            for (int g = 0; g < 8; ++g)
                s_hbuf[0][(mt1 * 16 + g + 8 * lh) * MIDW + col1] =
                    f2bf(ssp(acc[g] + bias1));
        }
        __syncthreads();
#pragma unroll 1
        for (int c = 0; c < NN; ++c) {
            const __bf16* hcur = s_hbuf[c & 1];
            __bf16*       hnxt = s_hbuf[(c + 1) & 1];
            // issue next chunk's global A-load early (hides under WMMA below)
            v16bf aN;
            if (c + 1 < NN)
                aN = load_a_f32(Eb + (c + 1) * NN * BASIS + mt1 * 16 * BASIS, BASIS, lane);
            if (c + 2 < NN)
                __builtin_prefetch(Eb + (c + 2) * NN * BASIS + tid * 4, 0, 1);
            // GEMM2 of current chunk + fused reduction against hx (mask j==c)
#pragma unroll
            for (int r = 0; r < 2; ++r) {
                v8f acc = zero8();
                acc = wmma_bf16(load_a_bf16(hcur + r * 16 * MIDW,      MIDW, lane), b2a, acc);
                acc = wmma_bf16(load_a_bf16(hcur + r * 16 * MIDW + 32, MIDW, lane), b2b, acc);
                float partial = 0.f;
#pragma unroll
                for (int g = 0; g < 8; ++g) {
                    int j = r * 16 + g + 8 * lh;
                    float h = (j == c) ? 0.f : s_hx[j * KE + col2];
                    partial += (acc[g] + bias2) * h;
                }
                atomicAdd(&s_z[c * KE + col2], partial);
            }
            // finish GEMM1 of next chunk into the other buffer
            if (c + 1 < NN) {
                v8f acc = wmma_bf16(aN, b1, zero8());
#pragma unroll
                for (int g = 0; g < 8; ++g)
                    hnxt[(mt1 * 16 + g + 8 * lh) * MIDW + col1] =
                        f2bf(ssp(acc[g] + bias1));
            }
            __syncthreads();
        }

        // -------- Phase D: x += ssp(z @ Wg1 + bg1) @ Wg2 + bg2 --------------
        {
            v16bf bw[4];
#pragma unroll
            for (int kt = 0; kt < 4; ++kt)
                bw[kt] = load_b_f32(wg1 + kt * 32 * KE + wv * 16, KE, lane);
#pragma unroll
            for (int r = 0; r < 2; ++r) {
                v8f acc = zero8();
#pragma unroll
                for (int kt = 0; kt < 4; ++kt)
                    acc = wmma_bf16(load_a_f32(s_z + r * 16 * KE + kt * 32, KE, lane),
                                    bw[kt], acc);
#pragma unroll
                for (int g = 0; g < 8; ++g)
                    s_g[(r * 16 + g + 8 * lh) * KE + col2] = f2bf(ssp(acc[g] + biag1));
            }
        }
        __syncthreads();
        {
            v16bf bw[4];
#pragma unroll
            for (int kt = 0; kt < 4; ++kt)
                bw[kt] = load_b_f32(wg2 + kt * 32 * KE + wv * 16, KE, lane);
#pragma unroll
            for (int r = 0; r < 2; ++r) {
                v8f acc = zero8();
#pragma unroll
                for (int kt = 0; kt < 4; ++kt)
                    acc = wmma_bf16(load_a_bf16(s_g + r * 16 * KE + kt * 32, KE, lane),
                                    bw[kt], acc);
#pragma unroll
                for (int g = 0; g < 8; ++g)
                    s_x[(r * 16 + g + 8 * lh) * KE + col2] += acc[g] + biag2;
            }
        }
        __syncthreads();
    }

    float* ob = out + (size_t)b * (NN * KE);
    for (int t = tid; t < NN * KE; t += 256) ob[t] = s_x[t];
}

extern "C" void kernel_launch(void* const* d_in, const int* in_sizes, int n_in,
                              void* d_out, int out_size, void* d_ws, size_t ws_size,
                              hipStream_t stream) {
    (void)in_sizes; (void)n_in; (void)out_size; (void)d_ws; (void)ws_size;
    ElectronicSchnet_76940044140594_kernel<<<dim3(BB), dim3(256), 0, stream>>>(
        (const float*)d_in[0],  (const float*)d_in[1],  (const float*)d_in[2],
        (const float*)d_in[3],  (const float*)d_in[4],  (const float*)d_in[5],
        (const float*)d_in[6],  (const float*)d_in[7],  (const float*)d_in[8],
        (const float*)d_in[9],  (const float*)d_in[10], (const float*)d_in[11],
        (const float*)d_in[12], (float*)d_out);
}
